// GNN_15401752723571
// MI455X (gfx1250) — matrix operand order repaired and checked by reading, not compiled, not run on hardware
//
#include <hip/hip_runtime.h>
#include <hip/hip_bf16.h>

// ---------------------------------------------------------------------------
// GCN (4x GCNConv + mean pool) for MI455X / gfx1250, wave32.
//   N = 100000 nodes, E = 3200000 edges, HID = 64, IN = 15, 128 graphs.
// Bandwidth-bound part: edge gather/scatter (fp32, atomic f32 adds, L2-resident
// accumulator). Compute part: h @ W via V_WMMA_F32_16X16X32_F16 (f16 inputs,
// fp32 accumulate) with the weight tile staged in LDS.
// ---------------------------------------------------------------------------

#define N_NODES   100000
#define HID       64
#define IN_DIM    15
#define NUM_GRAPHS 128

typedef __attribute__((ext_vector_type(16))) _Float16 v16h;
typedef __attribute__((ext_vector_type(8)))  float    v8f;

// ---------------- degree / norm ----------------

__global__ void init_deg_kernel(float* __restrict__ deg, int n) {
    int i = blockIdx.x * blockDim.x + threadIdx.x;
    if (i < n) deg[i] = 1.0f;               // self-loop contributes 1
}

__global__ void count_deg_kernel(const int* __restrict__ dst,
                                 float* __restrict__ deg, int nE) {
    int e = blockIdx.x * blockDim.x + threadIdx.x;
    if (e < nE) atomicAdd(&deg[dst[e]], 1.0f);
}

__global__ void calc_dis_kernel(const float* __restrict__ deg,
                                float* __restrict__ dis, int n) {
    int i = blockIdx.x * blockDim.x + threadIdx.x;
    if (i < n) dis[i] = rsqrtf(deg[i]);     // deg >= 1 always
}

// ---------------- layer 1 linear: x(15) @ w1(15x64) ----------------

__global__ void linear15_kernel(const float* __restrict__ x,
                                const float* __restrict__ w,
                                float* __restrict__ out, int n) {
    __shared__ float wl[IN_DIM * HID];
    for (int i = threadIdx.x; i < IN_DIM * HID; i += blockDim.x) wl[i] = w[i];
    __syncthreads();
    long long idx = (long long)blockIdx.x * blockDim.x + threadIdx.x;
    if (idx >= (long long)n * HID) return;
    long long node = idx >> 6;
    int d = (int)(idx & 63);
    const float* xr = x + node * IN_DIM;
    float s = 0.0f;
#pragma unroll
    for (int k = 0; k < IN_DIM; ++k) s = fmaf(xr[k], wl[k * HID + d], s);
    out[idx] = s;
}

// ---------------- WMMA GEMM: out = h(Nx64) @ W(64x64) ----------------
// Block: 128 threads = 4 waves. Each block computes a 16x64 output tile;
// wave w owns the 16-column slice [16w, 16w+16). Two K=32 WMMA steps.
__global__ void gcn_gemm_wmma_kernel(const float* __restrict__ h,
                                     const float* __restrict__ W,
                                     float* __restrict__ out) {
    __shared__ _Float16 Wlds[HID * HID];            // 8 KB f16 weight tile
    for (int i = threadIdx.x; i < HID * HID; i += blockDim.x)
        Wlds[i] = (_Float16)W[i];
    __syncthreads();

    const int wave    = threadIdx.x >> 5;           // 0..3
    const int lane    = threadIdx.x & 31;
    const int lane16  = lane & 15;
    const int halfSel = lane >> 4;                  // 0: lanes 0-15, 1: lanes 16-31
    const long long r0 = (long long)blockIdx.x * 16;
    const int n0 = wave * 16;

    const float* arow = h + (r0 + lane16) * HID;    // A row for this lane
    const int    col  = n0 + lane16;                // B column for this lane

    v8f acc = {};
#pragma unroll
    for (int c = 0; c < 2; ++c) {                   // K chunks: 0..31, 32..63
        // A fragment (16x32 f16): lanes 0-15 hold K base+{0..7,16..23},
        // lanes 16-31 hold K base+{8..15,24..31}  (base = 32c).
        const int abase = c * 32 + halfSel * 8;
        v16h a, b;
#pragma unroll
        for (int j = 0; j < 16; ++j) {
            int k = abase + (j < 8 ? j : j + 8);
            a[j] = (_Float16)arow[k];
        }
        // B fragment (32x16 f16): lane holds column `col`, same K order as A;
        // lanes 0-15 -> K base+{0..7,16..23}, lanes 16-31 -> K base+{8..15,24..31}.
        const int bbase = c * 32 + halfSel * 8;
#pragma unroll
        for (int j = 0; j < 16; ++j)
            b[j] = Wlds[(bbase + (j < 8 ? j : j + 8)) * HID + col];

        acc = __builtin_amdgcn_wmma_f32_16x16x32_f16(
            /*neg_a=*/false, a, /*neg_b=*/false, b,
            /*c_mod=*/(short)0, acc, /*reuse_a=*/false, /*reuse_b=*/false);
    }

    // D layout: VGPR v -> M = v + 8*halfSel, N = lane16.
    const long long rbase = r0 + halfSel * 8;
#pragma unroll
    for (int v = 0; v < 8; ++v)
        out[(rbase + v) * HID + n0 + lane16] = acc[v];
}

// ---------------- edge scatter: agg[dst] += hlin[src] * norm ----------------
// One wave32 per edge; each lane handles dims {lane, lane+32} (coalesced).
__global__ void scatter_kernel(const float* __restrict__ hlin,
                               const int* __restrict__ src,
                               const int* __restrict__ dst,
                               const float* __restrict__ dis,
                               float* __restrict__ agg, int nE) {
    long long e = (long long)blockIdx.x * (blockDim.x >> 5) + (threadIdx.x >> 5);
    if (e >= nE) return;
    const int lane = threadIdx.x & 31;
    const int s = src[e];
    const int d = dst[e];
    const float nrm = dis[s] * dis[d];
    const float* hs = hlin + (long long)s * HID;
    float*       ad = agg  + (long long)d * HID;
    atomicAdd(&ad[lane],      hs[lane]      * nrm);
    atomicAdd(&ad[lane + 32], hs[lane + 32] * nrm);
}

// ---------------- finalize: h = relu(agg + hlin*dis^2 (self-loop) + b) ------

__global__ void finalize_kernel(const float* __restrict__ agg,
                                const float* __restrict__ hlin,
                                const float* __restrict__ dis,
                                const float* __restrict__ bias,
                                float* __restrict__ h, int n) {
    long long idx = (long long)blockIdx.x * blockDim.x + threadIdx.x;
    if (idx >= (long long)n * HID) return;
    long long node = idx >> 6;
    int d = (int)(idx & 63);
    float di = dis[node];
    float v = agg[idx] + hlin[idx] * di * di + bias[d];
    h[idx] = v > 0.0f ? v : 0.0f;
}

// ---------------- pooling ----------------

__global__ void pool_sum_kernel(const float* __restrict__ h,
                                const int* __restrict__ batch,
                                float* __restrict__ sums,
                                float* __restrict__ cnt, int n) {
    long long idx = (long long)blockIdx.x * blockDim.x + threadIdx.x;
    if (idx >= (long long)n * HID) return;
    long long node = idx >> 6;
    int d = (int)(idx & 63);
    int b = batch[node];
    atomicAdd(&sums[b * HID + d], h[idx]);
    if (d == 0) atomicAdd(&cnt[b], 1.0f);
}

__global__ void pool_div_kernel(const float* __restrict__ sums,
                                const float* __restrict__ cnt,
                                float* __restrict__ out) {
    int i = blockIdx.x * blockDim.x + threadIdx.x;   // 128*64 threads exactly
    int b = i >> 6;
    out[i] = sums[i] / fmaxf(cnt[b], 1.0f);
}

// ---------------- host launch ----------------

static inline size_t align_up(size_t x) { return (x + 255) & ~(size_t)255; }

extern "C" void kernel_launch(void* const* d_in, const int* in_sizes, int n_in,
                              void* d_out, int out_size, void* d_ws, size_t ws_size,
                              hipStream_t stream) {
    (void)n_in; (void)out_size; (void)ws_size;

    const float* x   = (const float*)d_in[0];
    const float* w1  = (const float*)d_in[1];
    const float* b1  = (const float*)d_in[2];
    const float* w2  = (const float*)d_in[3];
    const float* b2  = (const float*)d_in[4];
    const float* w3  = (const float*)d_in[5];
    const float* b3  = (const float*)d_in[6];
    const float* w4  = (const float*)d_in[7];
    const float* b4  = (const float*)d_in[8];
    const int*   ei  = (const int*)d_in[9];
    const int*   bat = (const int*)d_in[10];

    const int n  = N_NODES;
    const int nE = in_sizes[9] / 2;
    const int* src = ei;
    const int* dst = ei + nE;

    // Workspace carve-out (~77 MB)
    char* ws = (char*)d_ws;
    size_t off = 0;
    const size_t szN  = align_up((size_t)n * sizeof(float));
    const size_t szNH = align_up((size_t)n * HID * sizeof(float));
    float* deg  = (float*)(ws + off); off += szN;
    float* dis  = (float*)(ws + off); off += szN;
    float* hlin = (float*)(ws + off); off += szNH;
    float* agg  = (float*)(ws + off); off += szNH;
    float* h    = (float*)(ws + off); off += szNH;
    float* sums = (float*)(ws + off); off += align_up(NUM_GRAPHS * HID * sizeof(float));
    float* cnt  = (float*)(ws + off); off += align_up(NUM_GRAPHS * sizeof(float));

    const int B = 256;
    const long long nh = (long long)n * HID;
    const int gridNH   = (int)((nh + B - 1) / B);
    const int gridN    = (n + B - 1) / B;
    const int gridE    = (nE + B - 1) / B;
    const int gridEw   = (nE + (B / 32) - 1) / (B / 32);   // one wave32 per edge
    const int gridGemm = n / 16;                           // 6250 row tiles

    // ---- gcn_norm ----
    init_deg_kernel<<<gridN, B, 0, stream>>>(deg, n);
    count_deg_kernel<<<gridE, B, 0, stream>>>(dst, deg, nE);
    calc_dis_kernel<<<gridN, B, 0, stream>>>(deg, dis, n);

    // ---- layer 1 (IN=15, plain FMA) ----
    linear15_kernel<<<gridNH, B, 0, stream>>>(x, w1, hlin, n);
    hipMemsetAsync(agg, 0, (size_t)n * HID * sizeof(float), stream);
    scatter_kernel<<<gridEw, B, 0, stream>>>(hlin, src, dst, dis, agg, nE);
    finalize_kernel<<<gridNH, B, 0, stream>>>(agg, hlin, dis, b1, h, n);

    // ---- layers 2..4 (WMMA GEMM) ----
    const float* Ws[3] = { w2, w3, w4 };
    const float* Bs[3] = { b2, b3, b4 };
    for (int l = 0; l < 3; ++l) {
        gcn_gemm_wmma_kernel<<<gridGemm, 128, 0, stream>>>(h, Ws[l], hlin);
        hipMemsetAsync(agg, 0, (size_t)n * HID * sizeof(float), stream);
        scatter_kernel<<<gridEw, B, 0, stream>>>(hlin, src, dst, dis, agg, nE);
        finalize_kernel<<<gridNH, B, 0, stream>>>(agg, hlin, dis, Bs[l], h, n);
    }

    // ---- mean pool ----
    hipMemsetAsync(sums, 0, NUM_GRAPHS * HID * sizeof(float), stream);
    hipMemsetAsync(cnt,  0, NUM_GRAPHS * sizeof(float), stream);
    pool_sum_kernel<<<gridNH, B, 0, stream>>>(h, bat, sums, cnt, n);
    pool_div_kernel<<<(NUM_GRAPHS * HID) / B, B, 0, stream>>>(sums, cnt, (float*)d_out);
}
